// dec_deeplabv3_contrast_68272800137382
// MI455X (gfx1250) — matrix operand order repaired and checked by reading, not compiled, not run on hardware
//
#include <hip/hip_runtime.h>
#include <hip/hip_bf16.h>
#include <math.h>

#define B_    4
#define CIN   2048
#define H_    65
#define W_    65
#define P_    (H_*W_)      /* 4225 */
#define CO_   256
#define CCAT  1280
#define NC_   19
#define QL_   2975
#define TEMP_ 0.2f
#define EPS_  1e-5f

typedef __attribute__((ext_vector_type(16))) __bf16 v16bf;
typedef __attribute__((ext_vector_type(8)))  __bf16 v8bf;
typedef __attribute__((ext_vector_type(8)))  float  v8f;

__device__ __forceinline__ __bf16 f2bf(float f) {
  unsigned u = __builtin_bit_cast(unsigned, f);
  unsigned r = u + 0x7FFFu + ((u >> 16) & 1u);   // round-to-nearest-even
  unsigned short h = (unsigned short)(r >> 16);
  return __builtin_bit_cast(__bf16, h);
}

__device__ __forceinline__ v8f wmma_bf16(v16bf a, v16bf b, v8f c) {
  return __builtin_amdgcn_wmma_f32_16x16x32_bf16(false, a, false, b, (short)0, c,
                                                 false, false);
}

// ---------------------------------------------------------------------------
// Implicit-GEMM conv: Y[n][co][p] += sum_{tap,k} Wt[tap][co][k] * X[n][k][p_shift]
// Block: 256 thr = 8 waves; macro tile 128co x 64px; wave tile 32co x 32px
// (2x2 register blocking -> 4 independent v_wmma per k-step).
// LDS B-tile stored [px][k] so each lane's B fragment is one 32-byte ds load.
// ---------------------------------------------------------------------------
__global__ void __launch_bounds__(256)
conv_wmma(const __bf16* __restrict__ X,   // [B][Kc][P]
          const __bf16* __restrict__ Wt,  // [taps][CO_][Kc]
          float* __restrict__ Y,          // [B][CO_][P]
          int Kc, int taps, int dil)
{
  __shared__ __align__(64) __bf16 Bt[64 * 32];   // [px][k], 4 KB

  const int tid  = threadIdx.x;
  const int lane = tid & 31;
  const int wave = tid >> 5;
  const int half = lane >> 4;      // K-half selector (this code's contiguous k = 16*half + s)
  const int lrow = lane & 15;      // A row (M) / B column (N)

  const int p0  = blockIdx.x * 64;                      // block pixel base
  const int cow = (blockIdx.y * 4 + (wave & 3)) * 32;   // wave co base (32 rows)
  const int pxw = (wave >> 2) * 32;                     // wave px base in tile
  const int n   = blockIdx.z;

  const __bf16* Xn = X + (size_t)n * Kc * P_;

  // cooperative loader: one pixel per thread, 8 K's per thread
  const int lpx = tid & 63;          // pixel within tile
  const int lkg = (tid >> 6) * 8;    // k group base 0/8/16/24
  const int pxg = p0 + lpx;
  const int hL = pxg / W_, wL = pxg % W_;

  v8f acc00 = {0,0,0,0,0,0,0,0}, acc01 = acc00, acc10 = acc00, acc11 = acc00;
  const __bf16 zb = f2bf(0.f);

  for (int tap = 0; tap < taps; ++tap) {
    const int dh = (taps == 9) ? ((tap / 3) - 1) * dil : 0;
    const int dw = (taps == 9) ? ((tap % 3) - 1) * dil : 0;

    const int h2 = hL + dh, w2 = wL + dw;
    const bool okL = (pxg < P_) &&
                     ((unsigned)h2 < (unsigned)H_) && ((unsigned)w2 < (unsigned)W_);
    const int p2 = h2 * W_ + w2;

    const __bf16* WtapA = Wt + (size_t)tap * CO_ * Kc;
    const __bf16* wrow0 = WtapA + (size_t)(cow + lrow) * Kc;
    const __bf16* wrow1 = WtapA + (size_t)(cow + 16 + lrow) * Kc;

    for (int k0 = 0; k0 < Kc; k0 += 32) {
      __syncthreads();
      {
        // stage B tile: thread writes 16 contiguous bytes of its pixel's k-run
        v8bf t;
        const __bf16* xs = Xn + (size_t)(k0 + lkg) * P_ + p2;
        #pragma unroll
        for (int i = 0; i < 8; ++i) t[i] = okL ? xs[(size_t)i * P_] : zb;
        *(v8bf*)(Bt + lpx * 32 + lkg) = t;
      }
      __syncthreads();

      // A fragments: contiguous K (half selects K 0-15|16-31), two aligned 16-byte loads each
      v16bf a0, a1;
      {
        v8bf lo = *(const v8bf*)(wrow0 + k0 + half * 16);
        v8bf hi = *(const v8bf*)(wrow0 + k0 + half * 16 + 8);
        #pragma unroll
        for (int i = 0; i < 8; ++i) { a0[i] = lo[i]; a0[8 + i] = hi[i]; }
      }
      {
        v8bf lo = *(const v8bf*)(wrow1 + k0 + half * 16);
        v8bf hi = *(const v8bf*)(wrow1 + k0 + half * 16 + 8);
        #pragma unroll
        for (int i = 0; i < 8; ++i) { a1[i] = lo[i]; a1[8 + i] = hi[i]; }
      }
      // B fragments: column = pixel, 32 contiguous bytes (half selects K 0-15|16-31)
      v16bf b0 = *(const v16bf*)(Bt + (pxw + lrow) * 32 + half * 16);
      v16bf b1 = *(const v16bf*)(Bt + (pxw + 16 + lrow) * 32 + half * 16);

      acc00 = wmma_bf16(a0, b0, acc00);
      acc01 = wmma_bf16(a0, b1, acc01);
      acc10 = wmma_bf16(a1, b0, acc10);
      acc11 = wmma_bf16(a1, b1, acc11);
    }
  }

  // D layout: VGPR r holds row (half*8 + r), column lrow
  {
    const int pc0 = p0 + pxw + lrow;
    const int pc1 = pc0 + 16;
    float* yb;
    if (pc0 < P_) {
      yb = Y + ((size_t)n * CO_ + cow + half * 8) * P_ + pc0;
      #pragma unroll
      for (int r = 0; r < 8; ++r) yb[(size_t)r * P_] = acc00[r];
      yb = Y + ((size_t)n * CO_ + cow + 16 + half * 8) * P_ + pc0;
      #pragma unroll
      for (int r = 0; r < 8; ++r) yb[(size_t)r * P_] = acc10[r];
    }
    if (pc1 < P_) {
      yb = Y + ((size_t)n * CO_ + cow + half * 8) * P_ + pc1;
      #pragma unroll
      for (int r = 0; r < 8; ++r) yb[(size_t)r * P_] = acc01[r];
      yb = Y + ((size_t)n * CO_ + cow + 16 + half * 8) * P_ + pc1;
      #pragma unroll
      for (int r = 0; r < 8; ++r) yb[(size_t)r * P_] = acc11[r];
    }
  }
}

// ---------------------------------------------------------------------------
// helpers / small kernels
// ---------------------------------------------------------------------------
__global__ void cvt_bf16(const float* __restrict__ s, __bf16* __restrict__ d, size_t n) {
  size_t i = (size_t)blockIdx.x * 256 + threadIdx.x;
  if (i < n) d[i] = f2bf(s[i]);
}

// src [CO][CI][taps] fp32 -> dst [tap][CO][CI] bf16
__global__ void repack_w(const float* __restrict__ src, __bf16* __restrict__ dst,
                         int CO, int CI, int taps) {
  size_t i = (size_t)blockIdx.x * 256 + threadIdx.x;
  size_t total = (size_t)CO * CI * taps;
  if (i >= total) return;
  int tap = (int)(i % taps);
  size_t oc = i / taps;                     // co*CI + ci
  dst[(size_t)tap * CO * CI + oc] = f2bf(src[oc * (size_t)taps + tap]);
}

__global__ void gp_mean(const float* __restrict__ x, float* __restrict__ gp) {
  __shared__ float red[256];
  const float* p = x + (size_t)blockIdx.x * P_;
  float s = 0.f;
  for (int i = threadIdx.x; i < P_; i += 256) s += p[i];
  red[threadIdx.x] = s; __syncthreads();
  for (int st = 128; st > 0; st >>= 1) {
    if (threadIdx.x < st) red[threadIdx.x] += red[threadIdx.x + st];
    __syncthreads();
  }
  if (threadIdx.x == 0) gp[blockIdx.x] = red[0] / (float)P_;
}

// 1x1 conv on pooled vector + BN over batch (N axis only, H=W=1) + ReLU
__global__ void gp_conv_bn(const float* __restrict__ wgp, const float* __restrict__ gp,
                           const float* __restrict__ g, const float* __restrict__ bb,
                           float* __restrict__ br0act) {
  int co = threadIdx.x;
  float y[B_];
  for (int n = 0; n < B_; ++n) {
    const float* w  = wgp + (size_t)co * CIN;
    const float* gn = gp + (size_t)n * CIN;
    float s = 0.f;
    for (int c = 0; c < CIN; ++c) s += w[c] * gn[c];
    y[n] = s;
  }
  float m = 0.f;
  for (int n = 0; n < B_; ++n) m += y[n];
  m *= (1.f / B_);
  float v = 0.f;
  for (int n = 0; n < B_; ++n) { float d = y[n] - m; v += d * d; }
  v *= (1.f / B_);
  float sc = g[co] * rsqrtf(v + EPS_);
  for (int n = 0; n < B_; ++n)
    br0act[n * CO_ + co] = fmaxf(sc * (y[n] - m) + bb[co], 0.f);
}

__global__ void fill_br0(const float* __restrict__ br0act, __bf16* __restrict__ cc) {
  size_t i = (size_t)blockIdx.x * 256 + threadIdx.x;
  if (i >= (size_t)B_ * CO_ * P_) return;
  int p = (int)(i % P_);
  size_t t = i / P_;
  int c = (int)(t % CO_);
  int n = (int)(t / CO_);
  cc[((size_t)n * CCAT + c) * P_ + p] = f2bf(br0act[n * CO_ + c]);
}

// per-channel mean/var over (N,H,W) -> scale/shift
__global__ void bn_stats(const float* __restrict__ y, const float* __restrict__ g,
                         const float* __restrict__ bb, float* __restrict__ scale,
                         float* __restrict__ shift) {
  __shared__ float r1[256], r2[256];
  int c = blockIdx.x;
  float s = 0.f, s2 = 0.f;
  for (int i = threadIdx.x; i < B_ * P_; i += 256) {
    int n = i / P_, p = i % P_;
    float v = y[((size_t)n * CO_ + c) * P_ + p];
    s += v; s2 += v * v;
  }
  r1[threadIdx.x] = s; r2[threadIdx.x] = s2; __syncthreads();
  for (int st = 128; st > 0; st >>= 1) {
    if (threadIdx.x < st) { r1[threadIdx.x] += r1[threadIdx.x + st]; r2[threadIdx.x] += r2[threadIdx.x + st]; }
    __syncthreads();
  }
  if (threadIdx.x == 0) {
    float mean = r1[0] / (float)(B_ * P_);
    float var  = r2[0] / (float)(B_ * P_) - mean * mean;
    float sc = g[c] * rsqrtf(var + EPS_);
    scale[c] = sc;
    shift[c] = bb[c] - mean * sc;
  }
}

__global__ void bn_apply(const float* __restrict__ y, const float* __restrict__ scale,
                         const float* __restrict__ shift, __bf16* __restrict__ cc,
                         int coff, float* __restrict__ fout) {
  size_t i = (size_t)blockIdx.x * 256 + threadIdx.x;
  if (i >= (size_t)B_ * CO_ * P_) return;
  int p = (int)(i % P_);
  size_t t = i / P_;
  int c = (int)(t % CO_);
  int n = (int)(t / CO_);
  float v = fmaxf(scale[c] * y[i] + shift[c], 0.f);
  if (cc)   cc[((size_t)n * CCAT + coff + c) * P_ + p] = f2bf(v);
  if (fout) fout[i] = v;
}

// 1x1 classifier (256->19) + bias, argmax -> pred, class counts
__global__ void fin_conv(const float* __restrict__ fea, const float* __restrict__ wfin,
                         const float* __restrict__ bfin, float* __restrict__ res,
                         int* __restrict__ pred, float* __restrict__ counts) {
  int i = blockIdx.x * 256 + threadIdx.x;
  if (i >= B_ * P_) return;
  int n = i / P_, p = i % P_;
  float acc[NC_];
  #pragma unroll
  for (int k = 0; k < NC_; ++k) acc[k] = bfin[k];
  const float* f = fea + (size_t)n * CO_ * P_ + p;
  for (int c = 0; c < CO_; ++c) {
    float v = f[(size_t)c * P_];
    #pragma unroll
    for (int k = 0; k < NC_; ++k) acc[k] += wfin[k * CO_ + c] * v;  // uniform w idx -> s_load
  }
  int best = 0; float bv = acc[0];
  #pragma unroll
  for (int k = 0; k < NC_; ++k) {
    res[((size_t)n * NC_ + k) * P_ + p] = acc[k];
    if (acc[k] > bv) { bv = acc[k]; best = k; }
  }
  pred[i] = best;
  atomicAdd(&counts[best], 1.f);
}

// csums[k][c] = sum over pixels with pred==k of fea[.,c,.]
__global__ void cls_sums(const float* __restrict__ fea, const int* __restrict__ pred,
                         float* __restrict__ csums) {
  __shared__ float red[256];
  int k = blockIdx.x / CO_;
  int c = blockIdx.x % CO_;
  float s = 0.f;
  for (int i = threadIdx.x; i < B_ * P_; i += 256) {
    if (pred[i] == k) {
      int n = i / P_, p = i % P_;
      s += fea[((size_t)n * CO_ + c) * P_ + p];
    }
  }
  red[threadIdx.x] = s; __syncthreads();
  for (int st = 128; st > 0; st >>= 1) {
    if (threadIdx.x < st) red[threadIdx.x] += red[threadIdx.x + st];
    __syncthreads();
  }
  if (threadIdx.x == 0) csums[k * CO_ + c] = red[0];
}

__global__ void mk_keys(const float* __restrict__ csums, const float* __restrict__ counts,
                        float* __restrict__ keys) {
  __shared__ float red[256];
  int k = blockIdx.x, c = threadIdx.x;
  float v = csums[k * CO_ + c] / fmaxf(counts[k], 1.f);
  red[c] = v * v; __syncthreads();
  for (int st = 128; st > 0; st >>= 1) {
    if (c < st) red[c] += red[c + st];
    __syncthreads();
  }
  float nrm = sqrtf(red[0]);
  keys[k * CO_ + c] = v / fmaxf(nrm, 1e-12f);
}

// per (k,j): streaming logsumexp over QL of (keys_k . queue_j,l)/(INNER*TEMP); stash diag row
__global__ void sims_lse(const float* __restrict__ keys, const float* __restrict__ queues,
                         float* __restrict__ lse_rows, float* __restrict__ diag) {
  __shared__ float kk[256];
  __shared__ float rm[256], rs[256];
  int k = blockIdx.x / NC_;
  int j = blockIdx.x % NC_;
  kk[threadIdx.x] = keys[k * CO_ + threadIdx.x];
  __syncthreads();
  const float* Q = queues + (size_t)j * CO_ * QL_;
  const float scl = 1.0f / ((float)CO_ * TEMP_);
  float tm = -INFINITY, ts = 0.f;
  for (int l = threadIdx.x; l < QL_; l += 256) {
    float d = 0.f;
    for (int c = 0; c < CO_; ++c) d += kk[c] * Q[(size_t)c * QL_ + l];
    float a = d * scl;
    if (k == j) diag[k * QL_ + l] = a;
    if (a > tm) { ts = ts * __expf(tm - a) + 1.f; tm = a; }
    else        { ts += __expf(a - tm); }
  }
  rm[threadIdx.x] = tm; rs[threadIdx.x] = ts; __syncthreads();
  for (int st = 128; st > 0; st >>= 1) {
    if (threadIdx.x < st) {
      float m1 = rm[threadIdx.x], s1 = rs[threadIdx.x];
      float m2 = rm[threadIdx.x + st], s2 = rs[threadIdx.x + st];
      float M = fmaxf(m1, m2);
      float S = ((m1 == -INFINITY) ? 0.f : s1 * __expf(m1 - M)) +
                ((m2 == -INFINITY) ? 0.f : s2 * __expf(m2 - M));
      rm[threadIdx.x] = M; rs[threadIdx.x] = S;
    }
    __syncthreads();
  }
  if (threadIdx.x == 0) lse_rows[k * NC_ + j] = rm[0] + logf(rs[0]);
}

__global__ void loss_k(const float* __restrict__ lse_rows, const float* __restrict__ diag,
                       const float* __restrict__ counts, float* __restrict__ out_loss) {
  __shared__ float red[256];
  float total = 0.f;
  for (int k = 0; k < NC_; ++k) {
    float m = -INFINITY;
    for (int j = 0; j < NC_; ++j) if (j != k) m = fmaxf(m, lse_rows[k * NC_ + j]);
    float s = 0.f;
    for (int j = 0; j < NC_; ++j) if (j != k) s += __expf(lse_rows[k * NC_ + j] - m);
    float lneg = m + logf(s);
    float part = 0.f;
    for (int l = threadIdx.x; l < QL_; l += 256) {
      float p = diag[k * QL_ + l];
      float M = fmaxf(p, lneg);
      part += (M + log1pf(__expf(-fabsf(p - lneg)))) - p;   // logaddexp(p,lneg)-p
    }
    red[threadIdx.x] = part; __syncthreads();
    for (int st = 128; st > 0; st >>= 1) {
      if (threadIdx.x < st) red[threadIdx.x] += red[threadIdx.x + st];
      __syncthreads();
    }
    if (counts[k] > 0.f) total += red[0] / (float)QL_;
    __syncthreads();
  }
  if (threadIdx.x == 0) *out_loss = total;
}

// ---------------------------------------------------------------------------
extern "C" void kernel_launch(void* const* d_in, const int* in_sizes, int n_in,
                              void* d_out, int out_size, void* d_ws, size_t ws_size,
                              hipStream_t stream) {
  (void)in_sizes; (void)n_in; (void)out_size; (void)ws_size;

  const float* x      = (const float*)d_in[0];
  const float* w_gp   = (const float*)d_in[1];
  const float* g_gp   = (const float*)d_in[2];
  const float* b_gp   = (const float*)d_in[3];
  const float* w_1x1  = (const float*)d_in[4];
  const float* g_1x1  = (const float*)d_in[5];
  const float* b_1x1  = (const float*)d_in[6];
  const float* w_d12  = (const float*)d_in[7];
  const float* g_d12  = (const float*)d_in[8];
  const float* b_d12  = (const float*)d_in[9];
  const float* w_d24  = (const float*)d_in[10];
  const float* g_d24  = (const float*)d_in[11];
  const float* b_d24  = (const float*)d_in[12];
  const float* w_d36  = (const float*)d_in[13];
  const float* g_d36  = (const float*)d_in[14];
  const float* b_d36  = (const float*)d_in[15];
  const float* w_head = (const float*)d_in[16];
  const float* g_head = (const float*)d_in[17];
  const float* b_head = (const float*)d_in[18];
  const float* w_fin  = (const float*)d_in[19];
  const float* b_fin  = (const float*)d_in[20];
  const float* queues = (const float*)d_in[21];

  char* ws = (char*)d_ws;
  size_t off = 0;
  auto alloc = [&](size_t bytes) -> char* {
    char* r = ws + off;
    off += (bytes + 255) & ~(size_t)255;
    return r;
  };

  __bf16* xb    = (__bf16*)alloc((size_t)B_ * CIN * P_ * 2);
  __bf16* wb1   = (__bf16*)alloc((size_t)CO_ * CIN * 2);
  __bf16* wb12  = (__bf16*)alloc((size_t)9 * CO_ * CIN * 2);
  __bf16* wb24  = (__bf16*)alloc((size_t)9 * CO_ * CIN * 2);
  __bf16* wb36  = (__bf16*)alloc((size_t)9 * CO_ * CIN * 2);
  __bf16* wbh   = (__bf16*)alloc((size_t)9 * CO_ * CCAT * 2);
  float*  gp    = (float*)alloc((size_t)B_ * CIN * 4);
  float*  br0a  = (float*)alloc((size_t)B_ * CO_ * 4);
  float*  ypre  = (float*)alloc((size_t)B_ * CO_ * P_ * 4);
  __bf16* cc    = (__bf16*)alloc((size_t)B_ * CCAT * P_ * 2);
  float*  fea   = (float*)alloc((size_t)B_ * CO_ * P_ * 4);
  float*  scale = (float*)alloc(CO_ * 4);
  float*  shift = (float*)alloc(CO_ * 4);
  int*    pred  = (int*)alloc((size_t)B_ * P_ * 4);
  float*  counts= (float*)alloc(NC_ * 4);
  float*  csums = (float*)alloc((size_t)NC_ * CO_ * 4);
  float*  keys  = (float*)alloc((size_t)NC_ * CO_ * 4);
  float*  lrows = (float*)alloc((size_t)NC_ * NC_ * 4);
  float*  diag  = (float*)alloc((size_t)NC_ * QL_ * 4);

  hipMemsetAsync(counts, 0, NC_ * sizeof(float), stream);

  // bf16 conversions / weight repacks
  size_t nx = (size_t)B_ * CIN * P_;
  cvt_bf16<<<(unsigned)((nx + 255) / 256), 256, 0, stream>>>(x, xb, nx);
  repack_w<<<(CO_ * CIN * 1 + 255) / 256, 256, 0, stream>>>(w_1x1, wb1, CO_, CIN, 1);
  repack_w<<<(CO_ * CIN * 9 + 255) / 256, 256, 0, stream>>>(w_d12, wb12, CO_, CIN, 9);
  repack_w<<<(CO_ * CIN * 9 + 255) / 256, 256, 0, stream>>>(w_d24, wb24, CO_, CIN, 9);
  repack_w<<<(CO_ * CIN * 9 + 255) / 256, 256, 0, stream>>>(w_d36, wb36, CO_, CIN, 9);
  repack_w<<<(CO_ * CCAT * 9 + 255) / 256, 256, 0, stream>>>(w_head, wbh, CO_, CCAT, 9);

  // global-pool branch
  gp_mean<<<B_ * CIN, 256, 0, stream>>>(x, gp);
  gp_conv_bn<<<1, 256, 0, stream>>>(w_gp, gp, g_gp, b_gp, br0a);
  size_t nb = (size_t)B_ * CO_ * P_;
  fill_br0<<<(unsigned)((nb + 255) / 256), 256, 0, stream>>>(br0a, cc);

  dim3 cgrid((P_ + 63) / 64, 2, B_);   // 67 pixel-tiles x 2 co-groups(128) x batch

  // branch 1: 1x1
  conv_wmma<<<cgrid, 256, 0, stream>>>(xb, wb1, ypre, CIN, 1, 1);
  bn_stats<<<CO_, 256, 0, stream>>>(ypre, g_1x1, b_1x1, scale, shift);
  bn_apply<<<(unsigned)((nb + 255) / 256), 256, 0, stream>>>(ypre, scale, shift, cc, 256, nullptr);
  // branch 2: dilated 12
  conv_wmma<<<cgrid, 256, 0, stream>>>(xb, wb12, ypre, CIN, 9, 12);
  bn_stats<<<CO_, 256, 0, stream>>>(ypre, g_d12, b_d12, scale, shift);
  bn_apply<<<(unsigned)((nb + 255) / 256), 256, 0, stream>>>(ypre, scale, shift, cc, 512, nullptr);
  // branch 3: dilated 24
  conv_wmma<<<cgrid, 256, 0, stream>>>(xb, wb24, ypre, CIN, 9, 24);
  bn_stats<<<CO_, 256, 0, stream>>>(ypre, g_d24, b_d24, scale, shift);
  bn_apply<<<(unsigned)((nb + 255) / 256), 256, 0, stream>>>(ypre, scale, shift, cc, 768, nullptr);
  // branch 4: dilated 36
  conv_wmma<<<cgrid, 256, 0, stream>>>(xb, wb36, ypre, CIN, 9, 36);
  bn_stats<<<CO_, 256, 0, stream>>>(ypre, g_d36, b_d36, scale, shift);
  bn_apply<<<(unsigned)((nb + 255) / 256), 256, 0, stream>>>(ypre, scale, shift, cc, 1024, nullptr);

  // head 3x3 over concat (K = 1280)
  conv_wmma<<<cgrid, 256, 0, stream>>>(cc, wbh, ypre, CCAT, 9, 1);
  bn_stats<<<CO_, 256, 0, stream>>>(ypre, g_head, b_head, scale, shift);
  bn_apply<<<(unsigned)((nb + 255) / 256), 256, 0, stream>>>(ypre, scale, shift, nullptr, 0, fea);

  // classifier + region stats + contrast loss
  float* res = (float*)d_out;
  float* out_loss = res + (size_t)B_ * NC_ * P_;
  fin_conv<<<(B_ * P_ + 255) / 256, 256, 0, stream>>>(fea, w_fin, b_fin, res, pred, counts);
  cls_sums<<<NC_ * CO_, 256, 0, stream>>>(fea, pred, csums);
  mk_keys<<<NC_, 256, 0, stream>>>(csums, counts, keys);
  sims_lse<<<NC_ * NC_, 256, 0, stream>>>(keys, queues, lrows, diag);
  loss_k<<<1, 256, 0, stream>>>(lrows, diag, counts, out_loss);
}
